// GraphClassifier_12489764897214
// MI455X (gfx1250) — matrix operand order repaired and checked by reading, not compiled, not run on hardware
//
#include <hip/hip_runtime.h>
#include <hip/hip_bf16.h>

// ---------------------------------------------------------------------------
// GraphClassifier for MI455X (gfx1250, wave32, WMMA + async-to-LDS DMA).
// HBM-bound (~0.95 GB streamed => ~40us floor at 23.3 TB/s). All GEMM-shaped
// work uses v_wmma_f32_16x16x32_bf16 (f32 accum). Tile traffic uses
// global_load_async_to_lds_b128, double-buffered with s_wait_asynccnt.
// Weights + encoder output are pre-converted to bf16 so the hot loops have
// (nearly) no conversion VALU: only the A-side x / masked-alpha fragments
// convert, via v_cvt_pk_bf16_f32.
// ---------------------------------------------------------------------------

typedef __attribute__((ext_vector_type(16))) __bf16 v16bf;
typedef __attribute__((ext_vector_type(8)))  __bf16 v8bf;
typedef __attribute__((ext_vector_type(8)))  float  v8f;

#define N1C 8192
#define N2C 4096
#define TILE_K 32

__device__ __forceinline__ unsigned short bfbits(float f) {
  union { float f; unsigned int u; } v; v.f = f;
  unsigned int u = v.u + 0x7FFFu + ((v.u >> 16) & 1u);   // RNE
  return (unsigned short)(u >> 16);
}

__device__ __forceinline__ __bf16 f2bf(float f) {
  unsigned short s = bfbits(f);
  __bf16 r; __builtin_memcpy(&r, &s, 2);
  return r;
}

// Packed f32x2 -> bf16x2 (v_cvt_pk_bf16_f32) with generic bit-cast handling.
__device__ __forceinline__ void cvt2(v16bf& d, int base, float x, float y) {
#if __has_builtin(__builtin_amdgcn_cvt_pk_bf16_f32)
  auto p = __builtin_amdgcn_cvt_pk_bf16_f32(x, y);
  unsigned u; __builtin_memcpy(&u, &p, 4);
  unsigned short l = (unsigned short)u, h = (unsigned short)(u >> 16);
  __bf16 bl, bh; __builtin_memcpy(&bl, &l, 2); __builtin_memcpy(&bh, &h, 2);
  d[base] = bl; d[base + 1] = bh;
#else
  d[base] = f2bf(x); d[base + 1] = f2bf(y);
#endif
}

__device__ __forceinline__ void pack8(v16bf& d, int base, float4 a, float4 b) {
  cvt2(d, base + 0, a.x, a.y);
  cvt2(d, base + 2, a.z, a.w);
  cvt2(d, base + 4, b.x, b.y);
  cvt2(d, base + 6, b.z, b.w);
}

// bf16 runs [p, p+8) and [p+16, p+24) in LDS -> v16bf (two aligned 16B loads, no conversion)
__device__ __forceinline__ v16bf cat8(const __bf16* p) {
  v8bf lo = *(const v8bf*)p;
  v8bf hi = *(const v8bf*)(p + 16);
  v16bf r;
  #pragma unroll
  for (int i = 0; i < 8; ++i) { r[i] = lo[i]; r[i + 8] = hi[i]; }
  return r;
}

// Low 32 bits of a flat shared-aperture address == LDS byte offset (ISA 10.2).
__device__ __forceinline__ unsigned ldsoff(const void* p) {
  return (unsigned)(size_t)p;
}

// Async DMA: 16B global -> LDS, tracked by ASYNCcnt (CDNA5 ISA).
__device__ __forceinline__ void async_g2l_b128(unsigned lds, const void* g) {
  asm volatile("global_load_async_to_lds_b128 %0, %1, off"
               :: "v"(lds), "v"(g) : "memory");
}
__device__ __forceinline__ void wait_async_5() {
  asm volatile("s_wait_asynccnt 5" ::: "memory");
}
__device__ __forceinline__ void wait_async_3() {
  asm volatile("s_wait_asynccnt 3" ::: "memory");
}
__device__ __forceinline__ void wait_async_0() {
  asm volatile("s_wait_asynccnt 0" ::: "memory");
}

// ---------------------------------------------------------------------------
// C[M,N] = A[M,K] @ Bw[N,K]^T + bias[N]   (A f32, Bw bf16 pre-converted)
// Block tile 128x64, 8 waves (wave w => rows [w*16,w*16+16) x 64 cols).
// ---------------------------------------------------------------------------
__global__ __launch_bounds__(256) void gemm_nt_bias(
    const float* __restrict__ A, int lda,
    const unsigned short* __restrict__ Bw, int ldb,   // bf16 bits [N,K]
    const float* __restrict__ bias,
    float* __restrict__ C, int ldc,
    int M, int N, int K)
{
  // pitches keep 16B async chunks aligned and skew LDS banks
  __shared__ __align__(16) float  sA[2][128][36];
  __shared__ __align__(16) __bf16 sB[2][64][40];

  const int tid  = threadIdx.x;
  const int wave = tid >> 5;
  const int lane = tid & 31;
  const int khi  = (lane >> 4) & 1;
  const int ln   = lane & 15;

  const int blockM = blockIdx.x * 128;
  const int blockN = blockIdx.y * 64;
  const int m0 = wave * 16;

  // 5 async b128 per wave per k-tile (A:4, B:1)
  auto issue = [&](int kb, int b) {
    #pragma unroll
    for (int j = 0; j < 4; ++j) {                 // A: 128 rows x 8 chunks f32
      int q = tid + 256 * j;
      int r = q >> 3, c = q & 7;
      async_g2l_b128(ldsoff(&sA[b][r][c * 4]),
                     A + (size_t)(blockM + r) * lda + kb + c * 4);
    }
    {                                             // B: 64 rows x 4 chunks bf16
      int r = tid >> 2, c = tid & 3;
      async_g2l_b128(ldsoff(&sB[b][r][c * 8]),
                     Bw + (size_t)(blockN + r) * ldb + kb + c * 8);
    }
  };

  v8f acc0 = {}, acc1 = {}, acc2 = {}, acc3 = {};

  const int nk = K / TILE_K;
  issue(0, 0);
  for (int t = 0; t < nk; ++t) {
    const int b = t & 1;
    if (t + 1 < nk) { issue((t + 1) * TILE_K, b ^ 1); wait_async_5(); }
    else            { wait_async_0(); }
    __syncthreads();

    // A fragment: K runs [khi*8, +8) and [16+khi*8, +8); cvt_pk to bf16
    const float* ra = &sA[b][m0 + ln][khi * 8];
    v16bf af;
    pack8(af, 0, *(const float4*)(ra),      *(const float4*)(ra + 4));
    pack8(af, 8, *(const float4*)(ra + 16), *(const float4*)(ra + 20));

    // B fragments: K runs [khi*8, +8) and [16+khi*8, +8) -- no conversion
    v16bf bf0 = cat8(&sB[b][ 0 + ln][khi * 8]);
    v16bf bf1 = cat8(&sB[b][16 + ln][khi * 8]);
    v16bf bf2 = cat8(&sB[b][32 + ln][khi * 8]);
    v16bf bf3 = cat8(&sB[b][48 + ln][khi * 8]);

    acc0 = __builtin_amdgcn_wmma_f32_16x16x32_bf16(false, af, false, bf0, (short)0, acc0, false, false);
    acc1 = __builtin_amdgcn_wmma_f32_16x16x32_bf16(false, af, false, bf1, (short)0, acc1, false, false);
    acc2 = __builtin_amdgcn_wmma_f32_16x16x32_bf16(false, af, false, bf2, (short)0, acc2, false, false);
    acc3 = __builtin_amdgcn_wmma_f32_16x16x32_bf16(false, af, false, bf3, (short)0, acc3, false, false);
    __syncthreads();
  }

  v8f* accs[4] = { &acc0, &acc1, &acc2, &acc3 };
  #pragma unroll
  for (int t = 0; t < 4; ++t) {
    int n = blockN + t * 16 + ln;
    float bv = bias[n];
    #pragma unroll
    for (int i = 0; i < 8; ++i) {
      int m = blockM + m0 + i + khi * 8;           // ISA f32 C/D layout
      C[(size_t)m * ldc + n] = (*accs[t])[i] + bv;
    }
  }
}

// ---------------------------------------------------------------------------
// Per-column batch statistics: mean + rsqrt(biased var + eps)
// ---------------------------------------------------------------------------
__global__ __launch_bounds__(256) void colstats(
    const float* __restrict__ H, int M, int N,
    float* __restrict__ mean, float* __restrict__ rstd)
{
  int n = blockIdx.x;
  float s = 0.f, s2 = 0.f;
  for (int r = threadIdx.x; r < M; r += 256) {
    float v = H[(size_t)r * N + n];
    s += v; s2 += v * v;
  }
  __shared__ float red[256], red2[256];
  red[threadIdx.x] = s; red2[threadIdx.x] = s2;
  __syncthreads();
  for (int o = 128; o > 0; o >>= 1) {
    if (threadIdx.x < o) {
      red[threadIdx.x]  += red[threadIdx.x + o];
      red2[threadIdx.x] += red2[threadIdx.x + o];
    }
    __syncthreads();
  }
  if (threadIdx.x == 0) {
    float mu  = red[0] / (float)M;
    float var = red2[0] / (float)M - mu * mu;
    mean[n] = mu;
    rstd[n] = rsqrtf(var + 1e-5f);
  }
}

__global__ __launch_bounds__(256) void bnrelu(
    float* __restrict__ H, int total, int N,
    const float* __restrict__ g, const float* __restrict__ b,
    const float* __restrict__ mean, const float* __restrict__ rstd)
{
  for (int idx = blockIdx.x * 256 + threadIdx.x; idx < total;
       idx += gridDim.x * 256) {
    int n = idx % N;
    float v = (H[idx] - mean[n]) * rstd[n] * g[n] + b[n];
    H[idx] = v > 0.f ? v : 0.f;
  }
}

// ---------------------------------------------------------------------------
// Elementwise f32 -> bf16 (weights)
// ---------------------------------------------------------------------------
__global__ __launch_bounds__(256) void f32_to_bf16(
    const float* __restrict__ src, unsigned short* __restrict__ dst, int n)
{
  for (int i = blockIdx.x * 256 + threadIdx.x; i < n; i += gridDim.x * 256)
    dst[i] = bfbits(src[i]);
}

// H[M,64] f32 -> HT[64][M] bf16 (transposed for attention B-side DMA)
__global__ __launch_bounds__(256) void h_to_bfT(
    const float* __restrict__ H, int M, unsigned short* __restrict__ HT)
{
  for (int idx = blockIdx.x * 256 + threadIdx.x; idx < M * 64;
       idx += gridDim.x * 256) {
    int r = idx >> 6, c = idx & 63;
    HT[(size_t)c * M + r] = bfbits(H[idx]);
  }
}

// ---------------------------------------------------------------------------
// deg[i] = sum_j adj[i][j]
// ---------------------------------------------------------------------------
__global__ __launch_bounds__(256) void rowdeg(
    const int* __restrict__ adj, int N, float* __restrict__ deg)
{
  int r = blockIdx.x;
  int s = 0;
  for (int c = threadIdx.x; c < N; c += 256) s += adj[(size_t)r * N + c];
  __shared__ int red[256];
  red[threadIdx.x] = s;
  __syncthreads();
  for (int o = 128; o > 0; o >>= 1) {
    if (threadIdx.x < o) red[threadIdx.x] += red[threadIdx.x + o];
    __syncthreads();
  }
  if (threadIdx.x == 0) deg[r] = (float)red[0];
}

// ---------------------------------------------------------------------------
// feat[m, 0:64] = ((adj .* alpha) @ H)[m] * W / deg[m] + H[m]
// Block tile 32x64; 8 waves = 2 row-tiles x 4 col-tiles (1 acc each).
// adj/alpha DMA'd raw f32/i32; B-side DMA'd from pre-transposed bf16 HT.
// ---------------------------------------------------------------------------
__global__ __launch_bounds__(256) void attn_agg(
    const int*   __restrict__ adj,   int lda_adj,
    const float* __restrict__ alpha, int lda_al,
    const unsigned short* __restrict__ HT,  // bf16 [64][Nrows]
    const float* __restrict__ H,            // f32  [Nrows][64] (residual)
    const float* __restrict__ deg,
    const float* __restrict__ Wp,           // scalar [1]
    float* __restrict__ feat,               // out (+ m*64 + n)
    int Nrows)
{
  __shared__ __align__(16) int    sJ[2][32][36];
  __shared__ __align__(16) float  sL[2][32][36];
  __shared__ __align__(16) __bf16 sB[2][64][40];

  const int tid  = threadIdx.x;
  const int wave = tid >> 5;
  const int lane = tid & 31;
  const int khi  = (lane >> 4) & 1;
  const int ln   = lane & 15;

  const int blockM = blockIdx.x * 32;
  const int m0 = (wave & 1) * 16;     // row tile
  const int nb = (wave >> 1) * 16;    // col tile
  const float Wv = Wp[0];

  // 3 async b128 per wave per k-tile (adj:1, alpha:1, B:1)
  auto issue = [&](int kb, int b) {
    {                                             // adj: 32 rows x 8 chunks
      int r = tid >> 3, c = tid & 7;
      async_g2l_b128(ldsoff(&sJ[b][r][c * 4]),
                     adj + (size_t)(blockM + r) * lda_adj + kb + c * 4);
    }
    {                                             // alpha: 32 rows x 8 chunks
      int r = tid >> 3, c = tid & 7;
      async_g2l_b128(ldsoff(&sL[b][r][c * 4]),
                     alpha + (size_t)(blockM + r) * lda_al + kb + c * 4);
    }
    {                                             // B: 64 rows x 4 chunks bf16
      int r = tid >> 2, c = tid & 3;
      async_g2l_b128(ldsoff(&sB[b][r][c * 8]),
                     HT + (size_t)r * Nrows + kb + c * 8);
    }
  };

  v8f acc = {};

  const int nk = Nrows / TILE_K;
  issue(0, 0);
  for (int t = 0; t < nk; ++t) {
    const int b = t & 1;
    if (t + 1 < nk) { issue((t + 1) * TILE_K, b ^ 1); wait_async_3(); }
    else            { wait_async_0(); }
    __syncthreads();

    // A fragment: masked alpha -> bf16 (mask in f32, then cvt_pk)
    const int*   rj = &sJ[b][m0 + ln][khi * 8];
    const float* rl = &sL[b][m0 + ln][khi * 8];
    int4   j0 = *(const int4*)(rj),        j1 = *(const int4*)(rj + 4);
    int4   j2 = *(const int4*)(rj + 16),   j3 = *(const int4*)(rj + 20);
    float4 l0 = *(const float4*)(rl),      l1 = *(const float4*)(rl + 4);
    float4 l2 = *(const float4*)(rl + 16), l3 = *(const float4*)(rl + 20);
    l0.x = j0.x ? l0.x : 0.f;  l0.y = j0.y ? l0.y : 0.f;
    l0.z = j0.z ? l0.z : 0.f;  l0.w = j0.w ? l0.w : 0.f;
    l1.x = j1.x ? l1.x : 0.f;  l1.y = j1.y ? l1.y : 0.f;
    l1.z = j1.z ? l1.z : 0.f;  l1.w = j1.w ? l1.w : 0.f;
    l2.x = j2.x ? l2.x : 0.f;  l2.y = j2.y ? l2.y : 0.f;
    l2.z = j2.z ? l2.z : 0.f;  l2.w = j2.w ? l2.w : 0.f;
    l3.x = j3.x ? l3.x : 0.f;  l3.y = j3.y ? l3.y : 0.f;
    l3.z = j3.z ? l3.z : 0.f;  l3.w = j3.w ? l3.w : 0.f;
    v16bf af;
    pack8(af, 0, l0, l1);
    pack8(af, 8, l2, l3);

    // B fragment: K runs [khi*8, +8) and [16+khi*8, +8) -- no conversion
    v16bf bfr = cat8(&sB[b][nb + ln][khi * 8]);

    acc = __builtin_amdgcn_wmma_f32_16x16x32_bf16(false, af, false, bfr, (short)0, acc, false, false);
    __syncthreads();
  }

  #pragma unroll
  for (int i = 0; i < 8; ++i) {
    int m = blockM + m0 + i + khi * 8;
    int n = nb + ln;
    float scale = Wv / deg[m];
    feat[(size_t)m * 64 + n] = acc[i] * scale + H[(size_t)m * 64 + n];
  }
}

// ---------------------------------------------------------------------------
// Classifier: logits = feat @ cw.T + cb ; softmax
// ---------------------------------------------------------------------------
__global__ void cls_init(float* __restrict__ logits, const float* __restrict__ cb) {
  if (threadIdx.x < 2) logits[threadIdx.x] = cb[threadIdx.x];
}

__global__ __launch_bounds__(256) void cls_dot(
    const float* __restrict__ feat, const float* __restrict__ cw,
    int L, float* __restrict__ logits)
{
  float s0 = 0.f, s1 = 0.f;
  for (int i = blockIdx.x * 256 + threadIdx.x; i < L; i += gridDim.x * 256) {
    float f = feat[i];
    s0 += f * cw[i];
    s1 += f * cw[(size_t)L + i];
  }
  __shared__ float r0[256], r1[256];
  r0[threadIdx.x] = s0; r1[threadIdx.x] = s1;
  __syncthreads();
  for (int o = 128; o > 0; o >>= 1) {
    if (threadIdx.x < o) {
      r0[threadIdx.x] += r0[threadIdx.x + o];
      r1[threadIdx.x] += r1[threadIdx.x + o];
    }
    __syncthreads();
  }
  if (threadIdx.x == 0) {
    atomicAdd(&logits[0], r0[0]);
    atomicAdd(&logits[1], r1[0]);
  }
}

__global__ void cls_softmax(const float* __restrict__ logits, float* __restrict__ out) {
  if (threadIdx.x == 0) {
    float a = logits[0], b = logits[1];
    float m = fmaxf(a, b);
    float ea = __expf(a - m), eb = __expf(b - m);
    float inv = 1.f / (ea + eb);
    out[0] = ea * inv;
    out[1] = eb * inv;
  }
}

// ---------------------------------------------------------------------------
// Host orchestration
// ---------------------------------------------------------------------------
static void run_encoder(const float* x, int M, int din,
                        const float* const* p,          // w1,b1,g1,be1,w2,...
                        unsigned short* wbf,            // bf16 weights scratch
                        float* buf0, float* buf1, float* mean, float* rstd,
                        hipStream_t stream)
{
  unsigned short* w1b = wbf;
  unsigned short* w2b = w1b + (size_t)256 * din;
  unsigned short* w3b = w2b + 128 * 256;
  f32_to_bf16<<<512, 256, 0, stream>>>(p[0], w1b, 256 * din);
  f32_to_bf16<<<64,  256, 0, stream>>>(p[4], w2b, 128 * 256);
  f32_to_bf16<<<16,  256, 0, stream>>>(p[8], w3b, 64 * 128);

  // Layer 1: [M,din] @ [256,din]^T -> buf0 [M,256]
  gemm_nt_bias<<<dim3(M / 128, 4), 256, 0, stream>>>(
      x, din, w1b, din, p[1], buf0, 256, M, 256, din);
  colstats<<<256, 256, 0, stream>>>(buf0, M, 256, mean, rstd);
  bnrelu<<<1024, 256, 0, stream>>>(buf0, M * 256, 256, p[2], p[3], mean, rstd);

  // Layer 2: [M,256] @ [128,256]^T -> buf1 [M,128]
  gemm_nt_bias<<<dim3(M / 128, 2), 256, 0, stream>>>(
      buf0, 256, w2b, 256, p[5], buf1, 128, M, 128, 256);
  colstats<<<128, 256, 0, stream>>>(buf1, M, 128, mean, rstd);
  bnrelu<<<1024, 256, 0, stream>>>(buf1, M * 128, 128, p[6], p[7], mean, rstd);

  // Layer 3: [M,128] @ [64,128]^T -> buf0 [M,64]
  gemm_nt_bias<<<dim3(M / 128, 1), 256, 0, stream>>>(
      buf1, 128, w3b, 128, p[9], buf0, 64, M, 64, 128);
  colstats<<<64, 256, 0, stream>>>(buf0, M, 64, mean, rstd);
  bnrelu<<<1024, 256, 0, stream>>>(buf0, M * 64, 64, p[10], p[11], mean, rstd);
}

extern "C" void kernel_launch(void* const* d_in, const int* in_sizes, int n_in,
                              void* d_out, int out_size, void* d_ws, size_t ws_size,
                              hipStream_t stream) {
  (void)in_sizes; (void)n_in; (void)out_size; (void)ws_size;

  const float* x1    = (const float*)d_in[0];
  const float* x2    = (const float*)d_in[1];
  const int*   adj1  = (const int*)  d_in[2];
  const int*   adj2  = (const int*)  d_in[3];
  const float* p1[12]; for (int i = 0; i < 12; ++i) p1[i] = (const float*)d_in[4 + i];
  const float* p2[12]; for (int i = 0; i < 12; ++i) p2[i] = (const float*)d_in[16 + i];
  const float* Wp     = (const float*)d_in[28];
  const float* alpha1 = (const float*)d_in[29];
  // d_in[30] (alpha2) unused: source bug reuses alpha1 for graph 2
  const float* cw     = (const float*)d_in[31];
  const float* cb     = (const float*)d_in[32];

  // Workspace layout (~25.4 MB)
  float* ws     = (float*)d_ws;
  float* buf0   = ws;                       // 8192*256 f32
  float* buf1   = buf0 + N1C * 256;         // 8192*256 f32
  float* mean   = buf1 + N1C * 256;         // 256
  float* rstd   = mean + 256;               // 256
  float* deg    = rstd + 256;               // 8192
  float* feat   = deg + N1C;                // (8192+4096)*64 f32
  float* logits = feat + (N1C + N2C) * 64;  // 2 (+4 pad = 24 B, not a 16B multiple)
  unsigned short* wbf  = (unsigned short*)(logits + 6);  // <= 2,138,112 bf16
  unsigned short* hbfT = wbf + 2200000;                  // 64*8192 bf16

  // ---- Graph 1 ----
  run_encoder(x1, N1C, N1C, p1, wbf, buf0, buf1, mean, rstd, stream);
  h_to_bfT<<<1024, 256, 0, stream>>>(buf0, N1C, hbfT);
  rowdeg<<<N1C, 256, 0, stream>>>(adj1, N1C, deg);
  attn_agg<<<N1C / 32, 256, 0, stream>>>(
      adj1, N1C, alpha1, N1C, hbfT, buf0, deg, Wp, feat, N1C);

  // ---- Graph 2 (alpha1[:N2,:N2] slice => row stride N1) ----
  run_encoder(x2, N2C, N2C, p2, wbf, buf0, buf1, mean, rstd, stream);
  h_to_bfT<<<1024, 256, 0, stream>>>(buf0, N2C, hbfT);
  rowdeg<<<N2C, 256, 0, stream>>>(adj2, N2C, deg);
  attn_agg<<<N2C / 32, 256, 0, stream>>>(
      adj2, N2C, alpha1, N1C, hbfT, buf0, deg, Wp, feat + (size_t)N1C * 64, N2C);

  // ---- Classifier ----
  cls_init<<<1, 32, 0, stream>>>(logits, cb);
  cls_dot<<<1024, 256, 0, stream>>>(feat, cw, (N1C + N2C) * 64, logits);
  cls_softmax<<<1, 32, 0, stream>>>(logits, (float*)d_out);
}
